// KimiDeltaAttention_5016521802517
// MI455X (gfx1250) — matrix operand order repaired and checked by reading, not compiled, not run on hardware
//
#include <hip/hip_runtime.h>
#include <math.h>

// ---- problem constants (match reference) ----
#define T_SEQ 1024
#define DMODEL 2048
#define NH 16
#define DH 128
#define KDIM 2048   // NH*DH
#define KC 4
#define EPSF 1e-6f

typedef __attribute__((ext_vector_type(16))) _Float16 v16h;
typedef __attribute__((ext_vector_type(8)))  float    v8f;
typedef __attribute__((ext_vector_type(4)))  int      v4i;

__device__ __forceinline__ float sigmf(float x) { return 1.f / (1.f + __expf(-x)); }

// gfx1250 async global->LDS DMA (ASYNCcnt) if the toolchain exposes it
#if __has_builtin(__builtin_amdgcn_global_load_async_to_lds_b128) && \
    __has_builtin(__builtin_amdgcn_s_wait_asynccnt)
#define USE_ASYNC_LDS 1
typedef __attribute__((address_space(1))) v4i* gv4i_p;   // global (AS1)
typedef __attribute__((address_space(3))) v4i* lv4i_p;   // LDS (AS3)
#else
#define USE_ASYNC_LDS 0
#endif

// =====================================================================
// WMMA GEMM: C[M,N] = A[M,K] @ W[K,N]; f32 memory, f16 multiply, f32 acc.
// Requirements: M % 128 == 0, K % 32 == 0 (true for all call sites).
// Block tile 128x128, K-step 32, 256 threads (8 waves in 4(M) x 2(N)).
// Tiles are staged RAW f32 into double-buffered LDS — via async DMA
// (global_load_async_to_lds_b128) on full-width blocks, synchronous
// global_load_b128 + ds_store otherwise — and converted to f16 at
// fragment-gather time.
// =====================================================================
#define BM 128
#define BN 128
#define BK 32
#define AP 36          // A-tile row stride in dwords (bank-conflict-free gather)
#define ASZ (BM * AP)  // 4608 dwords
#define BSZ (BK * BN)  // 4096 dwords

__global__ __launch_bounds__(256) void wmma_gemm_f16(
    const float* __restrict__ A, const float* __restrict__ W,
    float* __restrict__ C, int M, int N, int K)
{
  __shared__ float ldsA[2][ASZ];   // [m][k] f32, padded rows
  __shared__ float ldsB[2][BSZ];   // [k][n] f32, raw

  const int tid  = threadIdx.x;
  const int lane = tid & 31;
  const int wave = tid >> 5;
  const int wm   = wave & 3;     // 4 waves along M, 32 rows each
  const int wn   = wave >> 2;    // 2 waves along N, 64 cols each

  const int m0 = blockIdx.y * BM;
  const int n0 = blockIdx.x * BN;
  const bool fullN = (n0 + BN) <= N;

  const v8f vzero = {};
  v8f acc[2][4];
  #pragma unroll
  for (int i = 0; i < 2; ++i)
    #pragma unroll
    for (int j = 0; j < 4; ++j) acc[i][j] = vzero;

  // per-thread staging coordinates: 1024 x b128 transfers per tile, 4/thread
  const int ar = tid >> 3;            // A row 0..31 (+32*j)
  const int as = (tid & 7) * 4;       // A col dword 0..28
  const int bk = tid >> 5;            // B row 0..7 (+8*j)
  const int bs = (tid & 31) * 4;      // B col dword 0..124

  // ---- tile stager ----
  auto stage = [&](int buf, int k0) {
#if USE_ASYNC_LDS
    if (fullN) {
      #pragma unroll
      for (int j = 0; j < 4; ++j) {
        const int r = ar + 32 * j;
        __builtin_amdgcn_global_load_async_to_lds_b128(
            (gv4i_p)&A[(size_t)(m0 + r) * K + k0 + as],
            (lv4i_p)&ldsA[buf][r * AP + as], 0, 0);
        const int k = bk + 8 * j;
        __builtin_amdgcn_global_load_async_to_lds_b128(
            (gv4i_p)&W[(size_t)(k0 + k) * N + n0 + bs],
            (lv4i_p)&ldsB[buf][k * BN + bs], 0, 0);
      }
      return;
    }
#endif
    // synchronous fallback (also used for partial-N blocks)
    #pragma unroll
    for (int j = 0; j < 4; ++j) {
      const int r = ar + 32 * j;
      const float4 f = *(const float4*)&A[(size_t)(m0 + r) * K + k0 + as];
      *(float4*)&ldsA[buf][r * AP + as] = f;
      const int k = bk + 8 * j;
      if (fullN) {
        const float4 g = *(const float4*)&W[(size_t)(k0 + k) * N + n0 + bs];
        *(float4*)&ldsB[buf][k * BN + bs] = g;
      } else {
        #pragma unroll
        for (int i = 0; i < 4; ++i) {
          const int n = n0 + bs + i;
          ldsB[buf][k * BN + bs + i] = (n < N) ? W[(size_t)(k0 + k) * N + n] : 0.f;
        }
      }
    }
  };

  stage(0, 0);
  int buf = 0;

  for (int k0 = 0; k0 < K; k0 += BK) {
#if USE_ASYNC_LDS
    if (fullN) __builtin_amdgcn_s_wait_asynccnt(0);
#endif
    __syncthreads();                       // tile `buf` ready for all waves
    if (k0 + BK < K) stage(buf ^ 1, k0 + BK);

    // ---- gather fragments (f32 LDS -> f16 regs) per ISA layout ----
    // A 16x32: lanes 0-15 row M=lane, K {0-7,16-23}; lanes 16-31 K {8-15,24-31}
    const int akh = (lane >> 4) * 8;
    v16h afr[2], bfr[4];
    #pragma unroll
    for (int mi = 0; mi < 2; ++mi) {
      const int row = wm * 32 + mi * 16 + (lane & 15);
      const float* pa = &ldsA[buf][row * AP];
      const float4 a0 = *(const float4*)(pa + akh);
      const float4 a1 = *(const float4*)(pa + akh + 4);
      const float4 a2 = *(const float4*)(pa + akh + 16);
      const float4 a3 = *(const float4*)(pa + akh + 20);
      v16h f;
      f[0]=(_Float16)a0.x; f[1]=(_Float16)a0.y; f[2]=(_Float16)a0.z; f[3]=(_Float16)a0.w;
      f[4]=(_Float16)a1.x; f[5]=(_Float16)a1.y; f[6]=(_Float16)a1.z; f[7]=(_Float16)a1.w;
      f[8]=(_Float16)a2.x; f[9]=(_Float16)a2.y; f[10]=(_Float16)a2.z; f[11]=(_Float16)a2.w;
      f[12]=(_Float16)a3.x; f[13]=(_Float16)a3.y; f[14]=(_Float16)a3.z; f[15]=(_Float16)a3.w;
      afr[mi] = f;
    }
    // B 32x16: lanes 0-15 col N=lane%16, K {0-7,16-23}; lanes 16-31 K {8-15,24-31}.
    // Consecutive lanes read consecutive dwords -> conflict-free.
    const int bkb = (lane >> 4) * 8;
    #pragma unroll
    for (int ni = 0; ni < 4; ++ni) {
      const int col = wn * 64 + ni * 16 + (lane & 15);
      v16h f;
      #pragma unroll
      for (int kk = 0; kk < 16; ++kk)
        f[kk] = (_Float16)ldsB[buf][(bkb + (kk & 7) + ((kk >> 3) << 4)) * BN + col];
      bfr[ni] = f;
    }

    #pragma unroll
    for (int mi = 0; mi < 2; ++mi)
      #pragma unroll
      for (int ni = 0; ni < 4; ++ni)
        acc[mi][ni] = __builtin_amdgcn_wmma_f32_16x16x32_f16(
            false, afr[mi], false, bfr[ni], (short)0, acc[mi][ni], false, false);

    __syncthreads();                       // all waves done with tile `buf`
    buf ^= 1;
  }

  // ---- store C: VGPR r holds M=r (lanes 0-15) / M=r+8 (lanes 16-31), N=lane%16
  const int moff = (lane >= 16) ? 8 : 0;
  #pragma unroll
  for (int mi = 0; mi < 2; ++mi) {
    #pragma unroll
    for (int ni = 0; ni < 4; ++ni) {
      const int n = n0 + wn * 64 + ni * 16 + (lane & 15);
      if (n >= N) continue;
      #pragma unroll
      for (int r = 0; r < 8; ++r) {
        const int m = m0 + wm * 32 + mi * 16 + r + moff;
        C[(size_t)m * N + n] = acc[mi][ni][r];
      }
    }
  }
}

// =====================================================================
// Depthwise causal conv1d (K=4) + SiLU over [T, KDIM]
// =====================================================================
__global__ __launch_bounds__(256) void conv_silu_kernel(
    const float* __restrict__ in, const float* __restrict__ w,
    float* __restrict__ out, int total)
{
  const int i = blockIdx.x * blockDim.x + threadIdx.x;
  if (i >= total) return;
  const int c = i & (KDIM - 1);
  const int t = i >> 11;               // KDIM == 2048
  float acc = 0.f;
  #pragma unroll
  for (int j = 0; j < KC; ++j) {
    const int tt = t + j - (KC - 1);
    if (tt >= 0) acc = fmaf(w[j * KDIM + c], in[(size_t)tt * KDIM + c], acc);
  }
  out[i] = acc * sigmf(acc);
}

// g = -exp(A_log[h]) * softplus(g_raw + dt_bias[c])   (in place)
__global__ __launch_bounds__(256) void gate_kernel(
    float* __restrict__ g, const float* __restrict__ dt_bias,
    const float* __restrict__ A_log, int total)
{
  const int i = blockIdx.x * blockDim.x + threadIdx.x;
  if (i >= total) return;
  const int c = i & (KDIM - 1);
  const int h = c >> 7;                // DH == 128
  float x = g[i] + dt_bias[c];
  x = fminf(fmaxf(x, -20.f), 20.f);
  g[i] = -__expf(A_log[h]) * log1pf(__expf(x));
}

__global__ __launch_bounds__(256) void sigmoid_kernel(float* __restrict__ p, int total)
{
  const int i = blockIdx.x * blockDim.x + threadIdx.x;
  if (i < total) p[i] = sigmf(p[i]);
}

// per-(t,h) row of 128: p = p * rsqrt(sum(p^2)+eps) * scale   (wave per row)
__global__ __launch_bounds__(256) void l2norm_scale_kernel(
    float* __restrict__ p, int rows, float scale)
{
  const int wid  = blockIdx.x * (blockDim.x >> 5) + (threadIdx.x >> 5);
  const int lane = threadIdx.x & 31;
  if (wid >= rows) return;
  float* row = p + (size_t)wid * DH;
  float4 v = *(const float4*)(row + lane * 4);
  float ss = v.x * v.x + v.y * v.y + v.z * v.z + v.w * v.w;
  #pragma unroll
  for (int off = 16; off > 0; off >>= 1) ss += __shfl_xor(ss, off, 32);
  const float r = rsqrtf(ss + EPSF) * scale;
  v.x *= r; v.y *= r; v.z *= r; v.w *= r;
  *(float4*)(row + lane * 4) = v;
}

// gated per-head RMSNorm: o = o * rsqrt(mean(o^2)+eps) * norm_w * sigmoid(graw)
__global__ __launch_bounds__(256) void rmsnorm_gate_kernel(
    float* __restrict__ o, const float* __restrict__ graw,
    const float* __restrict__ norm_w, int rows)
{
  const int wid  = blockIdx.x * (blockDim.x >> 5) + (threadIdx.x >> 5);
  const int lane = threadIdx.x & 31;
  if (wid >= rows) return;
  float* row = o + (size_t)wid * DH;
  const float* grow = graw + (size_t)wid * DH;
  float4 v = *(const float4*)(row + lane * 4);
  float ss = v.x * v.x + v.y * v.y + v.z * v.z + v.w * v.w;
  #pragma unroll
  for (int off = 16; off > 0; off >>= 1) ss += __shfl_xor(ss, off, 32);
  const float rstd = rsqrtf(ss * (1.f / DH) + EPSF);
  const float4 g4 = *(const float4*)(grow + lane * 4);
  const int dv = lane * 4;
  float4 r;
  r.x = v.x * rstd * norm_w[dv + 0] * sigmf(g4.x);
  r.y = v.y * rstd * norm_w[dv + 1] * sigmf(g4.y);
  r.z = v.z * rstd * norm_w[dv + 2] * sigmf(g4.z);
  r.w = v.w * rstd * norm_w[dv + 3] * sigmf(g4.w);
  *(float4*)(row + lane * 4) = r;
}

// =====================================================================
// Gated delta-rule scan. One block per head, 512 threads (16 waves).
// Thread owns state column v = tid&127 over k-slice [32*(tid>>7), +32),
// 32 f32 state elements resident in VGPRs. Per-step vectors + cross-k
// reductions via LDS; per-wave LDS k-reads are uniform -> broadcast.
// =====================================================================
__global__ __launch_bounds__(512) void kda_scan_kernel(
    const float* __restrict__ qf, const float* __restrict__ kf,
    const float* __restrict__ v, const float* __restrict__ g,
    const float* __restrict__ beta, float* __restrict__ o)
{
  __shared__ float s_eg[DH], s_k[DH], s_q[DH], s_v[DH], s_u[DH];
  __shared__ float s_part[4][DH];
  __shared__ float s_beta;

  const int h   = blockIdx.x;
  const int tid = threadIdx.x;
  const int vi  = tid & (DH - 1);
  const int kb  = (tid >> 7) * 32;     // k-slice base, uniform per wave
  const int kg  = tid >> 7;

  float st[32];
  #pragma unroll
  for (int i = 0; i < 32; ++i) st[i] = 0.f;

  for (int t = 0; t < T_SEQ; ++t) {
    const size_t base = ((size_t)t * NH + h) * DH;
    if (tid < DH) {
      s_eg[tid] = __expf(g[base + tid]);
      s_k[tid]  = kf[base + tid];
    } else if (tid < 2 * DH) {
      const int i = tid - DH;
      s_q[i] = qf[base + i];
      s_v[i] = v[base + i];
    } else if (tid == 2 * DH) {
      s_beta = beta[(size_t)t * NH + h];
    }
    __syncthreads();                               // (1)

    // state decay + partial inner = k . state
    float partial = 0.f;
    #pragma unroll
    for (int i = 0; i < 32; ++i) {
      st[i] *= s_eg[kb + i];
      partial = fmaf(s_k[kb + i], st[i], partial);
    }
    s_part[kg][vi] = partial;
    __syncthreads();                               // (2)

    if (tid < DH) {
      const float inner = s_part[0][tid] + s_part[1][tid]
                        + s_part[2][tid] + s_part[3][tid];
      s_u[tid] = s_beta * (s_v[tid] - inner);
    }
    __syncthreads();                               // (3)

    // rank-1 update + partial o = q . state
    const float u = s_u[vi];
    float po = 0.f;
    #pragma unroll
    for (int i = 0; i < 32; ++i) {
      st[i] = fmaf(s_k[kb + i], u, st[i]);
      po = fmaf(s_q[kb + i], st[i], po);
    }
    s_part[kg][vi] = po;
    __syncthreads();                               // (4)

    if (tid < DH)
      o[base + tid] = s_part[0][tid] + s_part[1][tid]
                    + s_part[2][tid] + s_part[3][tid];
    // next-iter s_part writes are gated by barrier (1); no extra barrier needed
  }
}

// =====================================================================
// Launch
// =====================================================================
extern "C" void kernel_launch(void* const* d_in, const int* in_sizes, int n_in,
                              void* d_out, int out_size, void* d_ws, size_t ws_size,
                              hipStream_t stream)
{
  const float* x       = (const float*)d_in[0];
  const float* Wq      = (const float*)d_in[1];
  const float* Wk      = (const float*)d_in[2];
  const float* Wv      = (const float*)d_in[3];
  const float* conv_q  = (const float*)d_in[4];
  const float* conv_k  = (const float*)d_in[5];
  const float* conv_v  = (const float*)d_in[6];
  const float* Wfa     = (const float*)d_in[7];
  const float* Wfb     = (const float*)d_in[8];
  const float* dt_bias = (const float*)d_in[9];
  const float* A_log   = (const float*)d_in[10];
  const float* Wb      = (const float*)d_in[11];
  const float* Wga     = (const float*)d_in[12];
  const float* Wgb     = (const float*)d_in[13];
  const float* norm_w  = (const float*)d_in[14];
  const float* Wo      = (const float*)d_in[15];

  float* ws = (float*)d_ws;
  const size_t TK = (size_t)T_SEQ * KDIM;
  float* q_lin = ws; ws += TK;
  float* k_lin = ws; ws += TK;
  float* v_lin = ws; ws += TK;
  float* qpost = ws; ws += TK;
  float* kpost = ws; ws += TK;
  float* vpost = ws; ws += TK;
  float* gbuf  = ws; ws += TK;   // fa@Wfb, then gate in place
  float* goutb = ws; ws += TK;   // ga@Wgb (sigmoid applied in rmsnorm kernel)
  float* obuf  = ws; ws += TK;   // scan output, then rmsnorm in place
  float* fabuf = ws; ws += (size_t)T_SEQ * DH;
  float* gabuf = ws; ws += (size_t)T_SEQ * DH;
  float* betab = ws; ws += (size_t)T_SEQ * NH;

  const dim3 blk(256);
  auto gemm = [&](const float* A, const float* W, float* C, int M, int N, int K) {
    dim3 grid((N + BN - 1) / BN, (M + BM - 1) / BM);
    hipLaunchKernelGGL(wmma_gemm_f16, grid, blk, 0, stream, A, W, C, M, N, K);
  };

  // projections + gate inputs
  gemm(x, Wq,  q_lin, T_SEQ, KDIM, DMODEL);
  gemm(x, Wk,  k_lin, T_SEQ, KDIM, DMODEL);
  gemm(x, Wv,  v_lin, T_SEQ, KDIM, DMODEL);
  gemm(x, Wfa, fabuf, T_SEQ, DH,   DMODEL);
  gemm(x, Wga, gabuf, T_SEQ, DH,   DMODEL);
  gemm(x, Wb,  betab, T_SEQ, NH,   DMODEL);

  const int totTK = T_SEQ * KDIM;
  hipLaunchKernelGGL(conv_silu_kernel, dim3(totTK / 256), blk, 0, stream, q_lin, conv_q, qpost, totTK);
  hipLaunchKernelGGL(conv_silu_kernel, dim3(totTK / 256), blk, 0, stream, k_lin, conv_k, kpost, totTK);
  hipLaunchKernelGGL(conv_silu_kernel, dim3(totTK / 256), blk, 0, stream, v_lin, conv_v, vpost, totTK);

  gemm(fabuf, Wfb, gbuf,  T_SEQ, KDIM, DH);
  gemm(gabuf, Wgb, goutb, T_SEQ, KDIM, DH);

  hipLaunchKernelGGL(gate_kernel, dim3(totTK / 256), blk, 0, stream, gbuf, dt_bias, A_log, totTK);
  hipLaunchKernelGGL(sigmoid_kernel, dim3((T_SEQ * NH) / 256), blk, 0, stream, betab, T_SEQ * NH);

  const int rows = T_SEQ * NH;
  hipLaunchKernelGGL(l2norm_scale_kernel, dim3(rows / 8), blk, 0, stream, qpost, rows, 0.08838834764831845f); // DH^-0.5
  hipLaunchKernelGGL(l2norm_scale_kernel, dim3(rows / 8), blk, 0, stream, kpost, rows, 1.0f);

  hipLaunchKernelGGL(kda_scan_kernel, dim3(NH), dim3(512), 0, stream,
                     qpost, kpost, vpost, gbuf, betab, obuf);

  hipLaunchKernelGGL(rmsnorm_gate_kernel, dim3(rows / 8), blk, 0, stream, obuf, goutb, norm_w, rows);

  // out = o_n @ Wo  -> [T, DM] f32
  gemm(obuf, Wo, (float*)d_out, T_SEQ, DMODEL, KDIM);
}
